// IDCT_33466385170539
// MI455X (gfx1250) — matrix operand order repaired and checked by reading, not compiled, not run on hardware
//
#include <hip/hip_runtime.h>
#include <math.h>
#include <stdint.h>

// IDCT (DCT-III) as a WMMA GEMM: out = x @ B,  B[n][p] = w_n * cos(pi*n*(2p+1)/(2N))
// f16 hi/lo split (fp16x3) for near-fp32 accuracy on v_wmma_f32_16x16x32_f16.
// B tiles staged with gfx1250 async global->LDS copies (ASYNCcnt), A tiles through
// registers (f32 -> f16 hi/lo split), single barrier per K-step.

typedef __attribute__((ext_vector_type(16))) _Float16 v16h;
typedef __attribute__((ext_vector_type(8)))  _Float16 v8h;
typedef __attribute__((ext_vector_type(4)))  _Float16 v4h;
typedef __attribute__((ext_vector_type(8)))  float    v8f;

#define NDIM 4096
#define BM 128
#define BN 128
#define BK 32
#define KSTEPS (NDIM / BK)   // 128

// ---------------------------------------------------------------------------
// Precompute Bt[p][n] = w_n * cos(pi*n*(2p+1)/(2N)) as f16 hi + f16 lo.
// Integer phase reduction: cos(pi*m/(2N)) has period m mod 4N (power of two).
// ---------------------------------------------------------------------------
__global__ void idct_bt_precompute(_Float16* __restrict__ bth,
                                   _Float16* __restrict__ btl) {
  const int idx = blockIdx.x * blockDim.x + threadIdx.x;  // p*N + n
  const int p = idx >> 12;          // NDIM == 2^12
  const int n = idx & (NDIM - 1);
  const unsigned m = ((unsigned)n * (unsigned)(2 * p + 1)) & (4u * NDIM - 1u);
  const double ang = (double)m * (3.14159265358979323846 / (2.0 * (double)NDIM));
  float v = (float)cos(ang);
  if (n != 0) v *= 2.0f;
  const _Float16 hi = (_Float16)v;
  const _Float16 lo = (_Float16)(v - (float)hi);
  bth[idx] = hi;
  btl[idx] = lo;
}

// Async 16-byte global -> LDS copy (gfx1250, tracked by ASYNCcnt).
// GV addressing: vdst = per-lane LDS byte address, vaddr = 64-bit global address.
__device__ __forceinline__ void async_copy_b128(uint32_t lds_addr, const void* gptr) {
  const unsigned long long ga = (unsigned long long)(uintptr_t)gptr;
  asm volatile("global_load_async_to_lds_b128 %0, %1, off"
               :: "v"(lds_addr), "v"(ga)
               : "memory");
}
__device__ __forceinline__ void wait_async0() {
  asm volatile("s_wait_asynccnt 0x0" ::: "memory");
}

// ---------------------------------------------------------------------------
// Tiled WMMA GEMM, 128x128 block tile, BK=32, double-buffered LDS.
// ---------------------------------------------------------------------------
__global__ void __launch_bounds__(256)
idct_wmma_gemm(const float* __restrict__ x,
               const _Float16* __restrict__ bth,
               const _Float16* __restrict__ btl,
               float* __restrict__ out) {
  __shared__ __align__(16) _Float16 sAh[2][BM * BK];
  __shared__ __align__(16) _Float16 sAl[2][BM * BK];
  __shared__ __align__(16) _Float16 sBh[2][BN * BK];
  __shared__ __align__(16) _Float16 sBl[2][BN * BK];

  const int t     = threadIdx.x;
  const int blk_p = blockIdx.x;   // output-column block
  const int blk_m = blockIdx.y;   // output-row block

  // ---- A staging registers (global -> regs -> split -> LDS) ----
  float4 aReg[4];

  auto loadARegs = [&](int kt) {
    const int k0 = kt * BK;
    #pragma unroll
    for (int i = 0; i < 4; ++i) {
      const int li = t + i * 256;      // 0..1023
      const int r  = li >> 3;          // 0..127
      const int c4 = li & 7;           // 0..7
      aReg[i] = *(const float4*)(x + (size_t)(blk_m * BM + r) * NDIM + k0 + c4 * 4);
    }
  };

  auto prefetchA = [&](int kt) {
    const int li = t;                  // one line hint per thread
    const int r  = li >> 3;
    const int c4 = li & 7;
    __builtin_prefetch(x + (size_t)(blk_m * BM + r) * NDIM + kt * BK + c4 * 4, 0, 3);
  };

  auto storeLdsA = [&](int buf) {
    #pragma unroll
    for (int i = 0; i < 4; ++i) {
      const int li = t + i * 256;
      const int r  = li >> 3;
      const int c4 = li & 7;
      const float f[4] = {aReg[i].x, aReg[i].y, aReg[i].z, aReg[i].w};
      v4h h, l;
      #pragma unroll
      for (int j = 0; j < 4; ++j) {
        h[j] = (_Float16)f[j];
        l[j] = (_Float16)(f[j] - (float)h[j]);
      }
      *(v4h*)(&sAh[buf][r * BK + c4 * 4]) = h;
      *(v4h*)(&sAl[buf][r * BK + c4 * 4]) = l;
    }
  };

  // ---- B staging: async global -> LDS, 2x b128 per thread per array ----
  auto asyncB = [&](int kt, int buf) {
    const int k0 = kt * BK;
    const uint32_t ldsBh = (uint32_t)(uintptr_t)&sBh[buf][0];
    const uint32_t ldsBl = (uint32_t)(uintptr_t)&sBl[buf][0];
    #pragma unroll
    for (int i = 0; i < 2; ++i) {
      const int li = t + i * 256;      // 0..511 chunks of 8 halves
      const int r  = li >> 2;          // 0..127
      const int c8 = li & 3;           // 0..3
      const size_t g = (size_t)(blk_p * BN + r) * NDIM + k0 + c8 * 8;
      async_copy_b128(ldsBh + (uint32_t)li * 16u, bth + g);
      async_copy_b128(ldsBl + (uint32_t)li * 16u, btl + g);
    }
  };

  // ---- per-wave WMMA tiling: 8 waves in a 4(m) x 2(n) grid, 32x64 each ----
  const int lane  = t & 31;
  const int wid   = t >> 5;
  const int wm    = (wid & 3) * 32;   // row offset inside block tile
  const int wn    = (wid >> 2) * 64;  // col offset inside block tile
  const int lrow  = lane & 15;
  const int khalf = lane >> 4;

  // A operand (16x32 f16): lanes 0-15 rows m, K {0..7,16..23}; lanes 16-31 K {8..15,24..31}
  auto loadA = [&](const _Float16* base, int s) -> v16h {
    const _Float16* p0 = base + (wm + s * 16 + lrow) * BK + khalf * 8;
    v8h a0 = *(const v8h*)p0;
    v8h a1 = *(const v8h*)(p0 + 16);
    return __builtin_shufflevector(a0, a1, 0, 1, 2, 3, 4, 5, 6, 7,
                                           8, 9, 10, 11, 12, 13, 14, 15);
  };
  // B operand (32x16 f16), Bt stored row-per-output-col: lanes 0-15 K {0..7,16..23}, lanes 16-31 K {8..15,24..31}
  auto loadB = [&](const _Float16* base, int u) -> v16h {
    const _Float16* p0 = base + (wn + u * 16 + lrow) * BK + khalf * 8;
    v8h b0 = *(const v8h*)p0;
    v8h b1 = *(const v8h*)(p0 + 16);
    return __builtin_shufflevector(b0, b1, 0, 1, 2, 3, 4, 5, 6, 7,
                                           8, 9, 10, 11, 12, 13, 14, 15);
  };

  v8f acc[2][4];
  #pragma unroll
  for (int s = 0; s < 2; ++s)
    #pragma unroll
    for (int u = 0; u < 4; ++u)
      acc[s][u] = v8f{0.f, 0.f, 0.f, 0.f, 0.f, 0.f, 0.f, 0.f};

  auto compute = [&](int buf) {
    v16h ah[2], al[2];
    #pragma unroll
    for (int s = 0; s < 2; ++s) {
      ah[s] = loadA(sAh[buf], s);
      al[s] = loadA(sAl[buf], s);
    }
    #pragma unroll
    for (int u = 0; u < 4; ++u) {
      v16h bh = loadB(sBh[buf], u);
      v16h bl = loadB(sBl[buf], u);
      #pragma unroll
      for (int s = 0; s < 2; ++s) {
        acc[s][u] = __builtin_amdgcn_wmma_f32_16x16x32_f16(
            false, ah[s], false, bh, (short)0, acc[s][u], false, false);
        acc[s][u] = __builtin_amdgcn_wmma_f32_16x16x32_f16(
            false, al[s], false, bh, (short)0, acc[s][u], false, false);
        acc[s][u] = __builtin_amdgcn_wmma_f32_16x16x32_f16(
            false, ah[s], false, bl, (short)0, acc[s][u], false, false);
      }
    }
  };

  // ---- main K loop: single barrier per step, async B + reg-staged A ----
  loadARegs(0);
  asyncB(0, 0);
  storeLdsA(0);
  wait_async0();
  for (int kt = 0; kt < KSTEPS; ++kt) {
    const int buf = kt & 1;
    __syncthreads();                       // buf fully published; buf^1 free
    if (kt + 1 < KSTEPS) {
      loadARegs(kt + 1);                   // overlap with WMMA
      asyncB(kt + 1, buf ^ 1);             // async copy overlaps WMMA
    }
    if (kt + 2 < KSTEPS) prefetchA(kt + 2);
    compute(buf);
    if (kt + 1 < KSTEPS) {
      storeLdsA(buf ^ 1);                  // writes free buffer only
      wait_async0();                       // my async B done before next barrier
    }
  }

  // ---- epilogue: C/D layout -> global (lane: n=lane&15, m=v+8*(lane>>4)) ----
  // Non-temporal: out is write-once; keep L2 for x / Bt reuse.
  #pragma unroll
  for (int s = 0; s < 2; ++s) {
    #pragma unroll
    for (int u = 0; u < 4; ++u) {
      const int gm = blk_m * BM + wm + s * 16 + khalf * 8;
      const int gp = blk_p * BN + wn + u * 16 + lrow;
      float* o = out + (size_t)gm * NDIM + gp;
      #pragma unroll
      for (int v = 0; v < 8; ++v)
        __builtin_nontemporal_store(acc[s][u][v], o + (size_t)v * NDIM);
    }
  }
}

extern "C" void kernel_launch(void* const* d_in, const int* in_sizes, int n_in,
                              void* d_out, int out_size, void* d_ws, size_t ws_size,
                              hipStream_t stream) {
  const float* x = (const float*)d_in[0];
  float* out = (float*)d_out;

  // Workspace: Bt hi (32 MB) then Bt lo (32 MB)
  _Float16* bth = (_Float16*)d_ws;
  _Float16* btl = bth + (size_t)NDIM * NDIM;

  const int total = NDIM * NDIM;
  idct_bt_precompute<<<total / 256, 256, 0, stream>>>(bth, btl);

  dim3 grid(NDIM / BN, NDIM / BM);
  idct_wmma_gemm<<<grid, 256, 0, stream>>>(x, bth, btl, out);
}
